// CausalSelfAttention_8332236554516
// MI455X (gfx1250) — matrix operand order repaired and checked by reading, not compiled, not run on hardware
//
#include <hip/hip_runtime.h>
#include <math.h>

// ---------------- types ----------------
typedef _Float16 f16;
typedef __attribute__((ext_vector_type(16))) _Float16 v16h;
typedef __attribute__((ext_vector_type(8)))  _Float16 v8h;
typedef __attribute__((ext_vector_type(4)))  _Float16 v4h;
typedef __attribute__((ext_vector_type(8)))  float    v8f;

#define N_HEAD   16
#define N_KVHEAD 4
#define N_EMBD   2048
#define HEAD_DIM 128
#define SEQ_T    2048
#define BATCH    2
#define RMS_EPS  1.1920929e-07f

__device__ __forceinline__ v16h join16(v8h lo, v8h hi) {
  v16h r;
#pragma unroll
  for (int i = 0; i < 8; ++i) { r[i] = lo[i]; r[8 + i] = hi[i]; }
  return r;
}

__device__ __forceinline__ v8f wmma_f16(v16h a, v16h b, v8f c) {
  return __builtin_amdgcn_wmma_f32_16x16x32_f16(false, a, false, b, (short)0, c,
                                                false, false);
}

// async 16B global->LDS copy (CDNA5, ASYNCcnt-tracked, bypasses VGPRs)
__device__ __forceinline__ void async_cp16(unsigned lds_addr, const f16* gaddr) {
  asm volatile("global_load_async_to_lds_b128 %0, %1, off"
               :: "v"(lds_addr), "v"(gaddr) : "memory");
}

// ---------------- Kernel 0: f32 -> f16 conversion (4 elems/thread) ----------------
__global__ __launch_bounds__(256)
void cvt_f32_f16(const float* __restrict__ in, f16* __restrict__ out) {
  const size_t i = ((size_t)blockIdx.x * 256 + threadIdx.x) * 4;
  const float4 f = *(const float4*)(in + i);
  f16 o[4] __attribute__((aligned(8)));
  o[0] = (f16)f.x; o[1] = (f16)f.y; o[2] = (f16)f.z; o[3] = (f16)f.w;
  *(v4h*)(out + i) = *(v4h*)o;
}

// ---------------- Kernel 1: C[M,N] = A[M,K] * W[N,K]^T, f16 WMMA, async double-buffered ----------------
#define BM 128
#define BN 128
#define BK 32
#define LDT 40  // LDS row stride in halves (80B rows -> every 16B chunk aligned)

__global__ __launch_bounds__(256)
void gemm_f16_wmma_async(const f16* __restrict__ A, const f16* __restrict__ W,
                         float* __restrict__ C, int M, int N, int K) {
  __shared__ f16 As[2 * BM * LDT];
  __shared__ f16 Bs[2 * BN * LDT];

  const int tid  = threadIdx.x;
  const int lane = tid & 31;
  const int w    = tid >> 5;        // 0..7
  const int wm   = w >> 1;          // 0..3 (M dir)
  const int wn   = w & 1;           // 0..1 (N dir)
  const int lm   = lane & 15;
  const int hi   = (lane >> 4) & 1;
  const int bN   = blockIdx.x * BN;
  const int bM   = blockIdx.y * BM;

  v8f acc[2][4];
#pragma unroll
  for (int i = 0; i < 2; ++i)
#pragma unroll
    for (int j = 0; j < 4; ++j)
#pragma unroll
      for (int r = 0; r < 8; ++r) acc[i][j][r] = 0.0f;

  // staging: thread -> (row, 32B chunk-pair); 2 async B128 ops per matrix per tile
  const int ldRow = tid >> 1;        // 0..127
  const int ldCol = (tid & 1) * 16;  // halves: 0 or 16
  const f16* gA = A + (size_t)(bM + ldRow) * K + ldCol;
  const f16* gB = W + (size_t)(bN + ldRow) * K + ldCol;

  auto stage = [&](int kb, int ib) {
    const unsigned la =
        (unsigned)(uintptr_t)&As[ib * BM * LDT + ldRow * LDT + ldCol];
    const unsigned lb =
        (unsigned)(uintptr_t)&Bs[ib * BN * LDT + ldRow * LDT + ldCol];
    async_cp16(la,      gA + kb);
    async_cp16(la + 16, gA + kb + 8);
    async_cp16(lb,      gB + kb);
    async_cp16(lb + 16, gB + kb + 8);
  };

  stage(0, 0);  // prologue

  int ib = 0;
  for (int kb = 0; kb < K; kb += BK, ib ^= 1) {
    const bool more = (kb + BK) < K;
    if (more) {
      stage(kb + BK, ib ^ 1);  // prefetch next tile into other buffer
      asm volatile("s_wait_asynccnt 0x4" ::: "memory");  // current tile landed
    } else {
      asm volatile("s_wait_asynccnt 0x0" ::: "memory");
    }
    __syncthreads();  // all waves' copies visible

    const f16* Asb = &As[ib * BM * LDT];
    const f16* Bsb = &Bs[ib * BN * LDT];
    v16h af[2], bf[4];
    {
      const int kA = hi * 8;
#pragma unroll
      for (int i = 0; i < 2; ++i) {
        const f16* p = &Asb[(wm * 32 + i * 16 + lm) * LDT];
        af[i] = join16(*(const v8h*)(p + kA), *(const v8h*)(p + kA + 16));
      }
      const int kB = hi * 8;
#pragma unroll
      for (int j = 0; j < 4; ++j) {
        const f16* p = &Bsb[(wn * 64 + j * 16 + lm) * LDT + kB];
        bf[j] = join16(*(const v8h*)p, *(const v8h*)(p + 16));
      }
    }
#pragma unroll
    for (int i = 0; i < 2; ++i)
#pragma unroll
      for (int j = 0; j < 4; ++j)
        acc[i][j] = wmma_f16(af[i], bf[j], acc[i][j]);
    __syncthreads();  // reads done before buffer is re-staged
  }

#pragma unroll
  for (int i = 0; i < 2; ++i)
#pragma unroll
    for (int j = 0; j < 4; ++j)
#pragma unroll
      for (int r = 0; r < 8; ++r) {
        const int row = bM + wm * 32 + i * 16 + r + hi * 8;
        const int col = bN + wn * 64 + j * 16 + lm;
        C[(size_t)row * N + col] = acc[i][j][r];
      }
}

// ---------------- Kernel 2: RoPE + RMSNorm, f32 [B,T,H,D] -> f16 [B,H,T,D] ----------------
__global__ __launch_bounds__(256)
void rope_rms_kernel(const float* __restrict__ in, const float* __restrict__ cosp,
                     const float* __restrict__ sinp, f16* __restrict__ out, int H) {
  const int tid  = threadIdx.x;
  const int lane = tid & 31;
  const int w    = tid >> 5;
  const int row  = blockIdx.x * 8 + w;  // (b*T + t)*H + h
  const int h    = row % H;
  const int bt   = row / H;
  const int t    = bt % SEQ_T;
  const int b    = bt / SEQ_T;

  const float* src = in + (size_t)row * HEAD_DIM + lane * 4;
  const float* cp  = cosp + (size_t)t * HEAD_DIM + lane * 4;
  const float* sp  = sinp + (size_t)t * HEAD_DIM + lane * 4;

  float v[4], r[4];
  float ss = 0.0f;
#pragma unroll
  for (int i = 0; i < 4; ++i) v[i] = src[i];
#pragma unroll
  for (int i = 0; i < 4; ++i) {
    const float other = __shfl_xor(v[i], 16, 32);  // d <-> d +/- 64
    const float rot   = (lane < 16) ? -other : other;
    r[i] = v[i] * cp[i] + rot * sp[i];
    ss += r[i] * r[i];
  }
#pragma unroll
  for (int m = 1; m < 32; m <<= 1) ss += __shfl_xor(ss, m, 32);
  const float sc = rsqrtf(ss * (1.0f / (float)HEAD_DIM) + RMS_EPS);

  f16 o[4] __attribute__((aligned(8)));
#pragma unroll
  for (int i = 0; i < 4; ++i) o[i] = (f16)(r[i] * sc);
  f16* dst = out + ((((size_t)b * H + h) * SEQ_T + t) * HEAD_DIM) + lane * 4;
  *(v4h*)dst = *(v4h*)o;
}

// ---------------- Kernel 3: V f32 [B,T,KVH,D] -> f16 [B,KVH,D,T] ----------------
__global__ __launch_bounds__(256)
void v_transpose_f16(const float* __restrict__ v, f16* __restrict__ vt) {
  const int idx = blockIdx.x * blockDim.x + threadIdx.x;  // B*T*KVH*D
  const int d  = idx & (HEAD_DIM - 1);
  const int h  = (idx >> 7) & (N_KVHEAD - 1);
  const int bt = idx >> 9;
  const int t  = bt & (SEQ_T - 1);
  const int b  = bt >> 11;
  vt[((((size_t)b * N_KVHEAD + h) * HEAD_DIM + d) * SEQ_T) + t] = (f16)v[idx];
}

// ---------------- Kernel 4: causal flash attention (GQA), f16 output ----------------
// qh [B,16,T,128] f16; kh [B,4,T,128] f16; vt [B,4,128,T] f16; yo [B,T,2048] f16
__global__ __launch_bounds__(256)
void flash_attn_wmma(const f16* __restrict__ qh, const f16* __restrict__ kh,
                     const f16* __restrict__ vt, f16* __restrict__ yo) {
  __shared__ f16 Pl[8 * 16 * 32];  // per-wave P transpose scratch

  const int tid  = threadIdx.x;
  const int lane = tid & 31;
  const int w    = tid >> 5;
  const int lm   = lane & 15;
  const int hi   = (lane >> 4) & 1;

  const int blk  = blockIdx.x;          // B * 16 heads * (T/128)
  const int qblk = blk & 15;
  const int h    = (blk >> 4) & 15;
  const int b    = blk >> 8;
  const int kvh  = h >> 2;              // GQA group = 4
  const int qbase = qblk * 128 + w * 16;

  const f16* qp = qh + (((size_t)b * N_HEAD + h) * SEQ_T) * HEAD_DIM;
  const f16* kp = kh + (((size_t)b * N_KVHEAD + kvh) * SEQ_T) * HEAD_DIM;
  const f16* vp = vt + (((size_t)b * N_KVHEAD + kvh) * HEAD_DIM) * SEQ_T;

  // Q fragments for the 4 d-chunks of 32
  v16h qf[4];
  {
    const f16* qrow = qp + (size_t)(qbase + lm) * HEAD_DIM;
#pragma unroll
    for (int c = 0; c < 4; ++c) {
      const int d0 = c * 32 + hi * 8;
      qf[c] = join16(*(const v8h*)(qrow + d0), *(const v8h*)(qrow + d0 + 16));
    }
  }

  v8f o[8];
#pragma unroll
  for (int c = 0; c < 8; ++c)
#pragma unroll
    for (int r = 0; r < 8; ++r) o[c][r] = 0.0f;
  float mx[8], l[8];
#pragma unroll
  for (int r = 0; r < 8; ++r) { mx[r] = -INFINITY; l[r] = 0.0f; }

  const float scale = 0.08838834764831845f;  // 1/sqrt(128)
  const int nkb = (qbase + 15) / 32 + 1;     // causal key-block bound

  for (int kb = 0; kb < nkb; ++kb) {
    const int n0 = kb * 32;
    v8f s0, s1;
#pragma unroll
    for (int r = 0; r < 8; ++r) { s0[r] = 0.0f; s1[r] = 0.0f; }

#pragma unroll
    for (int c = 0; c < 4; ++c) {
      const int dB = c * 32 + hi * 8;
      const f16* k0 = kp + (size_t)(n0 + lm) * HEAD_DIM + dB;
      const f16* k1 = kp + (size_t)(n0 + 16 + lm) * HEAD_DIM + dB;
      const v16h b0 = join16(*(const v8h*)k0, *(const v8h*)(k0 + 16));
      const v16h b1 = join16(*(const v8h*)k1, *(const v8h*)(k1 + 16));
      s0 = wmma_f16(qf[c], b0, s0);
      s1 = wmma_f16(qf[c], b1, s1);
    }

    // online softmax (per-row stats live in 16 lanes of each half-wave)
#pragma unroll
    for (int r = 0; r < 8; ++r) {
      const int qrow = qbase + r + hi * 8;
      float x0 = (n0 + lm      <= qrow) ? s0[r] * scale : -INFINITY;
      float x1 = (n0 + 16 + lm <= qrow) ? s1[r] * scale : -INFINITY;
      float rmax = fmaxf(x0, x1);
#pragma unroll
      for (int m = 1; m < 16; m <<= 1) rmax = fmaxf(rmax, __shfl_xor(rmax, m, 32));
      const float mnew  = fmaxf(mx[r], rmax);
      const float alpha = __expf(mx[r] - mnew);
      const float p0 = __expf(x0 - mnew);
      const float p1 = __expf(x1 - mnew);
      float rsum = p0 + p1;
#pragma unroll
      for (int m = 1; m < 16; m <<= 1) rsum += __shfl_xor(rsum, m, 32);
      l[r]  = l[r] * alpha + rsum;
      mx[r] = mnew;
#pragma unroll
      for (int c = 0; c < 8; ++c) o[c][r] *= alpha;
      const int m_ = r + hi * 8;
      Pl[(w * 16 + m_) * 32 + lm]      = (f16)p0;
      Pl[(w * 16 + m_) * 32 + 16 + lm] = (f16)p1;
    }
    asm volatile("" ::: "memory");  // keep DS stores before DS reloads

    // reload P as an A-fragment (16 x 32 over keys)
    v16h pa;
    {
      const f16* pp = &Pl[(w * 16 + lm) * 32 + hi * 8];
      pa = join16(*(const v8h*)pp, *(const v8h*)(pp + 16));
    }
#pragma unroll
    for (int c = 0; c < 8; ++c) {
      const f16* vv = vp + (size_t)(c * 16 + lm) * SEQ_T + n0 + hi * 8;
      const v16h bv = join16(*(const v8h*)vv, *(const v8h*)(vv + 16));
      o[c] = wmma_f16(pa, bv, o[c]);
    }
  }

#pragma unroll
  for (int r = 0; r < 8; ++r) {
    const float inv = 1.0f / l[r];
    const int t = qbase + r + hi * 8;
    f16* dst = yo + ((size_t)(b * SEQ_T + t) * N_EMBD) + h * HEAD_DIM;
#pragma unroll
    for (int c = 0; c < 8; ++c) dst[c * 16 + lm] = (f16)(o[c][r] * inv);
  }
}

// ---------------- host launch ----------------
extern "C" void kernel_launch(void* const* d_in, const int* in_sizes, int n_in,
                              void* d_out, int out_size, void* d_ws, size_t ws_size,
                              hipStream_t stream) {
  const float* x     = (const float*)d_in[0];
  const float* cosp  = (const float*)d_in[1];
  const float* sinp  = (const float*)d_in[2];
  const float* wq    = (const float*)d_in[3];
  const float* wk    = (const float*)d_in[4];
  const float* wv    = (const float*)d_in[5];
  const float* wproj = (const float*)d_in[6];
  float* out = (float*)d_out;

  const int M = BATCH * SEQ_T;  // 4096
  const int K = N_EMBD;         // 2048
  const int NKV = N_KVHEAD * HEAD_DIM;  // 512

  char* ws = (char*)d_ws;
  size_t off = 0;
  f16*   xh    = (f16*)(ws + off);   off += (size_t)M * N_EMBD * sizeof(f16);       // 16MB
  f16*   wqh   = (f16*)(ws + off);   off += (size_t)N_EMBD * N_EMBD * sizeof(f16);  //  8MB
  f16*   wkh   = (f16*)(ws + off);   off += (size_t)NKV * N_EMBD * sizeof(f16);     //  2MB
  f16*   wvh   = (f16*)(ws + off);   off += (size_t)NKV * N_EMBD * sizeof(f16);     //  2MB
  f16*   wph   = (f16*)(ws + off);   off += (size_t)N_EMBD * N_EMBD * sizeof(f16);  //  8MB
  float* qf32  = (float*)(ws + off); off += (size_t)M * N_EMBD * sizeof(float);     // 32MB
  float* kf32  = (float*)(ws + off); off += (size_t)M * NKV * sizeof(float);        //  8MB
  float* vf32  = (float*)(ws + off); off += (size_t)M * NKV * sizeof(float);        //  8MB
  f16*   qh    = (f16*)(ws + off);   off += (size_t)M * N_EMBD * sizeof(f16);       // 16MB
  f16*   kh    = (f16*)(ws + off);   off += (size_t)M * NKV * sizeof(f16);          //  4MB
  f16*   vth   = (f16*)(ws + off);   off += (size_t)M * NKV * sizeof(f16);          //  4MB
  f16*   yatth = (f16*)(ws + off);   off += (size_t)M * N_EMBD * sizeof(f16);       // 16MB

  // 0) one-shot f32 -> f16 of activations and weights
  cvt_f32_f16<<<(M * N_EMBD) / 1024, 256, 0, stream>>>(x, xh);
  cvt_f32_f16<<<(N_EMBD * N_EMBD) / 1024, 256, 0, stream>>>(wq, wqh);
  cvt_f32_f16<<<(NKV * N_EMBD) / 1024, 256, 0, stream>>>(wk, wkh);
  cvt_f32_f16<<<(NKV * N_EMBD) / 1024, 256, 0, stream>>>(wv, wvh);
  cvt_f32_f16<<<(N_EMBD * N_EMBD) / 1024, 256, 0, stream>>>(wproj, wph);

  // 1) QKV projections (x @ W^T), async-staged WMMA
  gemm_f16_wmma_async<<<dim3(N_EMBD / BN, M / BM), 256, 0, stream>>>(xh, wqh, qf32, M, N_EMBD, K);
  gemm_f16_wmma_async<<<dim3(NKV / BN, M / BM), 256, 0, stream>>>(xh, wkh, kf32, M, NKV, K);
  gemm_f16_wmma_async<<<dim3(NKV / BN, M / BM), 256, 0, stream>>>(xh, wvh, vf32, M, NKV, K);

  // 2) RoPE + RMSNorm (+transpose to [B,H,T,D] f16)
  rope_rms_kernel<<<(BATCH * SEQ_T * N_HEAD) / 8, 256, 0, stream>>>(qf32, cosp, sinp, qh, N_HEAD);
  rope_rms_kernel<<<(BATCH * SEQ_T * N_KVHEAD) / 8, 256, 0, stream>>>(kf32, cosp, sinp, kh, N_KVHEAD);

  // 3) V -> f16 transposed [B,KVH,D,T]
  v_transpose_f16<<<(BATCH * SEQ_T * NKV / HEAD_DIM), 256, 0, stream>>>(vf32, vth);

  // 4) causal flash attention (writes f16)
  flash_attn_wmma<<<BATCH * N_HEAD * (SEQ_T / 128), 256, 0, stream>>>(qh, kh, vth, yatth);

  // 5) output projection
  gemm_f16_wmma_async<<<dim3(N_EMBD / BN, M / BM), 256, 0, stream>>>(yatth, wph, out, M, N_EMBD, K);
}
